// FeedForwardSAT_65231963291790
// MI455X (gfx1250) — matrix operand order repaired and checked by reading, not compiled, not run on hardware
//
#include <hip/hip_runtime.h>

#define FDIM 256
#define ROUNDS 16
#define N_VARS 50000
#define N_CLAUSES 210000
#define N_LITS (2 * N_VARS)
#define LN_EPS 1e-3f

typedef __attribute__((ext_vector_type(16))) __bf16 v16bf;
typedef __attribute__((ext_vector_type(8)))  __bf16 v8bf;
typedef __attribute__((ext_vector_type(4)))  __bf16 v4bf;
typedef __attribute__((ext_vector_type(8)))  float  v8f;

__device__ __forceinline__ __bf16 f2bf(float f) { return (__bf16)f; }

// ---------------------------------------------------------------------------
// gfx1250 async global->LDS copy (ASYNCcnt-tracked DMA), per CDNA5 ISA
// ---------------------------------------------------------------------------
__device__ __forceinline__ void async_cp16(unsigned lds_off, const void* gptr) {
  asm volatile("global_load_async_to_lds_b128 %0, %1, off"
               :: "v"(lds_off), "v"(gptr)
               : "memory");
}
__device__ __forceinline__ void wait_async_le8() {
  asm volatile("s_wait_asynccnt 0x8" ::: "memory");
}
__device__ __forceinline__ void wait_async_0() {
  asm volatile("s_wait_asynccnt 0x0" ::: "memory");
}
__device__ __forceinline__ unsigned lds_off32(const void* p) {
  // LDS aperture: flat addr low 32 bits are the workgroup-relative LDS address
  return (unsigned)(uintptr_t)p;
}

// ---------------------------------------------------------------------------
// Utility kernels
// ---------------------------------------------------------------------------
__global__ void zero_f32_kernel(float4* p, long n4) {
  long i = (long)blockIdx.x * blockDim.x + threadIdx.x;
  if (i < n4) p[i] = make_float4(0.f, 0.f, 0.f, 0.f);
}

__global__ void init_state_kernel(float* st, const float* __restrict__ init, int nrows) {
  long i = (long)blockIdx.x * blockDim.x + threadIdx.x;
  long total = (long)nrows * FDIM;
  if (i < total) st[i] = init[i & (FDIM - 1)] * 0.0625f;  // 1/sqrt(256)
}

// W[K][N] f32 row-major  ->  WT[N][K] bf16 (B fragments contiguous in K)
__global__ void convert_wt_kernel(const float* __restrict__ W, __bf16* __restrict__ WT,
                                  int K, int N) {
  long i = (long)blockIdx.x * blockDim.x + threadIdx.x;
  if (i < (long)K * N) {
    int k = (int)(i / N), n = (int)(i % N);
    WT[(long)n * K + k] = f2bf(W[i]);
  }
}

// dst[didx[e]] += src[sidx[e]]  (row-wise, F=256 floats). 64 float4-chunks/edge.
__global__ void scatter_add_kernel(const float* __restrict__ src, const int* __restrict__ sidx,
                                   float* __restrict__ dst, const int* __restrict__ didx,
                                   int nnz) {
  long t = (long)blockIdx.x * blockDim.x + threadIdx.x;
  int e = (int)(t >> 6);
  int j = (int)(t & 63);
  if (e >= nnz) return;
  int s = sidx[e], d = didx[e];
  float4 v = ((const float4*)(src + (long)s * FDIM))[j];
  float* p = dst + (long)d * FDIM + j * 4;
  atomicAdd(p + 0, v.x);
  atomicAdd(p + 1, v.y);
  atomicAdd(p + 2, v.z);
  atomicAdd(p + 3, v.w);
}

// ---------------------------------------------------------------------------
// Block-cooperative GEMM with async-DMA'd, double-buffered weight slices.
// One wave computes a 16-row x 256-col f32 tile; the 4 waves of a block share
// each 32x256 bf16 weight K-slice staged in LDS in WMMA-B layout [N=256][K=32].
// Per-thread DMA: 8 x b128 chunks per slice (4 waves x 8 lanes-worth = 16 KB).
// ---------------------------------------------------------------------------
__device__ __forceinline__ void issue_slice(const __bf16* __restrict__ WT, int kd, int kb,
                                            __bf16* dstbuf, int tid) {
#pragma unroll
  for (int i = 0; i < 8; ++i) {
    int c = i * 128 + tid;          // 0..1023 chunk id
    int n = c >> 2;                 // output column 0..255
    int part = c & 3;               // 16B chunk within the 64B K-row
    const __bf16* g = WT + (long)n * kd + kb + part * 8;
    async_cp16(lds_off32(dstbuf + n * 32 + part * 8), g);
  }
}

template<int KD>
__device__ __forceinline__ void block_gemm(const __bf16* __restrict__ Awave,
                                           const __bf16* __restrict__ WT,
                                           __bf16* wbuf /* 2 x 8192 bf16 */,
                                           v8f acc[16], int tid) {
  const int lane = tid & 31;
  const int hf = lane >> 4;
  const int lr = lane & 15;
  constexpr int T = KD / 32;

  issue_slice(WT, KD, 0, wbuf, tid);  // prologue: slice 0 -> buf0
  for (int t = 0; t < T; ++t) {
    if (t + 1 < T) {
      issue_slice(WT, KD, (t + 1) * 32, wbuf + ((t + 1) & 1) * 8192, tid);
      wait_async_le8();               // slice t complete (in-order), t+1 in flight
    } else {
      wait_async_0();
    }
    __syncthreads();                  // slice t visible to all 4 waves

    const __bf16* B = wbuf + (t & 1) * 8192;
    const __bf16* ap = Awave + (long)lr * 512 + t * 32 + hf * 16;
    v8bf lo = *(const v8bf*)ap;
    v8bf hi = *(const v8bf*)(ap + 8);
    v16bf af = __builtin_shufflevector(lo, hi, 0, 1, 2, 3, 4, 5, 6, 7,
                                       8, 9, 10, 11, 12, 13, 14, 15);
#pragma unroll
    for (int n = 0; n < 16; ++n) {
      v16bf bfv = *(const v16bf*)(B + (n * 16 + lr) * 32 + hf * 16);
      acc[n] = __builtin_amdgcn_wmma_f32_16x16x32_bf16(
          false, af, false, bfv, (short)0, acc[n], false, false);
    }
    __syncthreads();                  // all waves done reading before overwrite
  }
}

// bias + ReLU, write tile back to LDS as bf16 (next layer's A), zero acc.
__device__ __forceinline__ void relu_bias_to_lds(v8f acc[16], const float* __restrict__ bias,
                                                 __bf16* __restrict__ A, int lane) {
  const int hf = lane >> 4, lr = lane & 15;
#pragma unroll
  for (int n = 0; n < 16; ++n) {
    float b = bias[n * 16 + lr];
#pragma unroll
    for (int v = 0; v < 8; ++v) {
      float x = fmaxf(acc[n][v] + b, 0.0f);
      A[(long)(v + hf * 8) * 512 + n * 16 + lr] = f2bf(x);
      acc[n][v] = 0.0f;
    }
  }
}

__device__ __forceinline__ void stage_rows(__bf16* A, const float* __restrict__ in0,
                                           const float* __restrict__ in1,
                                           int rowBase, int N, int lane) {
  for (int r = 0; r < 16; ++r) {
    int row = rowBase + r;
    row = row < N ? row : N - 1;
    const float4* p0 = (const float4*)(in0 + (long)row * FDIM);
    const float4* p1 = (const float4*)(in1 + (long)row * FDIM);
#pragma unroll
    for (int c = 0; c < 2; ++c) {
      int j = lane + c * 32;
      float4 a = p0[j];
      v4bf pa = {f2bf(a.x), f2bf(a.y), f2bf(a.z), f2bf(a.w)};
      *(v4bf*)(A + r * 512 + j * 4) = pa;
      float4 bq = p1[j];
      v4bf pb = {f2bf(bq.x), f2bf(bq.y), f2bf(bq.z), f2bf(bq.w)};
      *(v4bf*)(A + r * 512 + 256 + j * 4) = pb;
    }
  }
}

// ---------------------------------------------------------------------------
// Fused update MLP: x = LN( W3·relu(W2·relu(W1·[in0|in1] + b1) + b2) + b3 )
// ---------------------------------------------------------------------------
__global__ __launch_bounds__(128) void mlp_ln_kernel(
    const float* __restrict__ in0, const float* __restrict__ in1,
    const __bf16* __restrict__ WT1, const float* __restrict__ b1,
    const __bf16* __restrict__ WT2, const float* __restrict__ b2,
    const __bf16* __restrict__ WT3, const float* __restrict__ b3,
    const float* __restrict__ gamma, const float* __restrict__ beta,
    float* __restrict__ out, int N) {
  __shared__ __attribute__((aligned(32))) __bf16 act[4 * 16 * 512];  // 64 KB
  __shared__ __attribute__((aligned(32))) __bf16 wbuf[2 * 8192];     // 32 KB dbl-buf
  const int tid = threadIdx.x;
  const int wave = tid >> 5;
  const int lane = tid & 31;
  const int hf = lane >> 4, lr = lane & 15;
  const int rowBase = (blockIdx.x * 4 + wave) * 16;
  __bf16* A = act + wave * 16 * 512;

  stage_rows(A, in0, in1, rowBase, N, lane);
  __syncthreads();

  const v8f vzero = {0.f, 0.f, 0.f, 0.f, 0.f, 0.f, 0.f, 0.f};
  v8f acc[16];
#pragma unroll
  for (int n = 0; n < 16; ++n) acc[n] = vzero;

  block_gemm<512>(A, WT1, wbuf, acc, tid);
  relu_bias_to_lds(acc, b1, A, lane);
  __syncthreads();
  block_gemm<256>(A, WT2, wbuf, acc, tid);
  relu_bias_to_lds(acc, b2, A, lane);
  __syncthreads();
  block_gemm<256>(A, WT3, wbuf, acc, tid);

  // bias3 + LayerNorm (in-register, shfl_xor across the 16-lane half)
#pragma unroll
  for (int n = 0; n < 16; ++n) {
    float b = b3[n * 16 + lr];
#pragma unroll
    for (int v = 0; v < 8; ++v) acc[n][v] += b;
  }
#pragma unroll
  for (int v = 0; v < 8; ++v) {
    float s = 0.f, s2 = 0.f;
#pragma unroll
    for (int n = 0; n < 16; ++n) {
      float x = acc[n][v];
      s += x;
      s2 += x * x;
    }
#pragma unroll
    for (int off = 1; off < 16; off <<= 1) {
      s += __shfl_xor(s, off, 32);
      s2 += __shfl_xor(s2, off, 32);
    }
    float mu = s * (1.0f / FDIM);
    float var = s2 * (1.0f / FDIM) - mu * mu;
    float rstd = rsqrtf(var + LN_EPS);
    int row = rowBase + v + hf * 8;
    if (row < N) {
#pragma unroll
      for (int n = 0; n < 16; ++n) {
        int col = n * 16 + lr;
        out[(long)row * FDIM + col] = (acc[n][v] - mu) * rstd * gamma[col] + beta[col];
      }
    }
  }
}

// ---------------------------------------------------------------------------
// Vote MLP: [lits_pos | lits_neg] (512) -> 256 relu -> 256 relu -> dot(256) + b
// ---------------------------------------------------------------------------
__global__ __launch_bounds__(128) void vote_kernel(
    const float* __restrict__ lits,
    const __bf16* __restrict__ WT1, const float* __restrict__ b1,
    const __bf16* __restrict__ WT2, const float* __restrict__ b2,
    const float* __restrict__ W3, const float* __restrict__ b3,
    float* __restrict__ out, int N) {
  __shared__ __attribute__((aligned(32))) __bf16 act[4 * 16 * 512];
  __shared__ __attribute__((aligned(32))) __bf16 wbuf[2 * 8192];
  const int tid = threadIdx.x;
  const int wave = tid >> 5;
  const int lane = tid & 31;
  const int hf = lane >> 4, lr = lane & 15;
  const int rowBase = (blockIdx.x * 4 + wave) * 16;
  __bf16* A = act + wave * 16 * 512;

  {
    int rbase = rowBase;
    for (int r = 0; r < 16; ++r) {
      int row = rbase + r;
      row = row < N ? row : N - 1;
      const float4* p0 = (const float4*)(lits + (long)row * FDIM);
      const float4* p1 = (const float4*)(lits + (long)(N_VARS + row) * FDIM);
#pragma unroll
      for (int c = 0; c < 2; ++c) {
        int j = lane + c * 32;
        float4 a = p0[j];
        v4bf pa = {f2bf(a.x), f2bf(a.y), f2bf(a.z), f2bf(a.w)};
        *(v4bf*)(A + r * 512 + j * 4) = pa;
        float4 bq = p1[j];
        v4bf pb = {f2bf(bq.x), f2bf(bq.y), f2bf(bq.z), f2bf(bq.w)};
        *(v4bf*)(A + r * 512 + 256 + j * 4) = pb;
      }
    }
  }
  __syncthreads();

  const v8f vzero = {0.f, 0.f, 0.f, 0.f, 0.f, 0.f, 0.f, 0.f};
  v8f acc[16];
#pragma unroll
  for (int n = 0; n < 16; ++n) acc[n] = vzero;

  block_gemm<512>(A, WT1, wbuf, acc, tid);
  relu_bias_to_lds(acc, b1, A, lane);
  __syncthreads();
  block_gemm<256>(A, WT2, wbuf, acc, tid);

  // bias2 + relu in registers
#pragma unroll
  for (int n = 0; n < 16; ++n) {
    float b = b2[n * 16 + lr];
#pragma unroll
    for (int v = 0; v < 8; ++v) acc[n][v] = fmaxf(acc[n][v] + b, 0.0f);
  }
  // final dot with W3[256] (f32) + b3
  float part[8];
#pragma unroll
  for (int v = 0; v < 8; ++v) part[v] = 0.f;
#pragma unroll
  for (int n = 0; n < 16; ++n) {
    float w = W3[n * 16 + lr];
#pragma unroll
    for (int v = 0; v < 8; ++v) part[v] += acc[n][v] * w;
  }
  float bb = b3[0];
#pragma unroll
  for (int v = 0; v < 8; ++v) {
    float s = part[v];
#pragma unroll
    for (int off = 1; off < 16; off <<= 1) s += __shfl_xor(s, off, 32);
    int row = rowBase + v + hf * 8;
    if (lr == 0 && row < N) out[row] = s + bb;
  }
}

// ---------------------------------------------------------------------------
extern "C" void kernel_launch(void* const* d_in, const int* in_sizes, int n_in,
                              void* d_out, int out_size, void* d_ws, size_t ws_size,
                              hipStream_t stream) {
  // --- input index mapping (insertion-order vs sorted-pytree flattening) ---
  int iLinit, iCinit, iCW[3], iCb[3], iLW[3], iLb[3], iVW[3], iVb[3];
  int iLg, iLbe, iCg, iCbe, iLit, iCls;
  if (in_sizes[0] == 256) {  // insertion order: params first (L_init is leaf 0)
    iLinit = 0; iCinit = 1;
    iCW[0] = 2;  iCb[0] = 3;  iCW[1] = 4;  iCb[1] = 5;  iCW[2] = 6;  iCb[2] = 7;
    iLW[0] = 8;  iLb[0] = 9;  iLW[1] = 10; iLb[1] = 11; iLW[2] = 12; iLb[2] = 13;
    iVW[0] = 14; iVb[0] = 15; iVW[1] = 16; iVb[1] = 17; iVW[2] = 18; iVb[2] = 19;
    iLg = 20; iLbe = 21; iCg = 22; iCbe = 23; iLit = 24; iCls = 25;
  } else {  // alphabetical pytree order: clause_idx, lit_idx, n_clauses, n_vars, params...
    iCls = 0; iLit = 1;
    iCbe = 4; iCg = 5; iCinit = 6;
    iCW[0] = 7;  iCb[0] = 8;  iCW[1] = 9;  iCb[1] = 10; iCW[2] = 11; iCb[2] = 12;
    iLbe = 13; iLg = 14; iLinit = 15;
    iLW[0] = 16; iLb[0] = 17; iLW[1] = 18; iLb[1] = 19; iLW[2] = 20; iLb[2] = 21;
    iVW[0] = 22; iVb[0] = 23; iVW[1] = 24; iVb[1] = 25; iVW[2] = 26; iVb[2] = 27;
  }
  const int nnz = in_sizes[iLit];
  const int* lit_idx = (const int*)d_in[iLit];
  const int* clause_idx = (const int*)d_in[iCls];

  // --- workspace layout ---
  char* ws = (char*)d_ws;
  size_t off = 0;
  auto wsalloc = [&](size_t bytes) -> void* {
    void* p = ws + off;
    off += (bytes + 255) & ~(size_t)255;
    return p;
  };
  float* lits = (float*)wsalloc((size_t)N_LITS * FDIM * 4);
  float* cls  = (float*)wsalloc((size_t)N_CLAUSES * FDIM * 4);
  float* msg  = (float*)wsalloc((size_t)N_CLAUSES * FDIM * 4);
  __bf16* CT1 = (__bf16*)wsalloc((size_t)512 * 256 * 2);
  __bf16* CT2 = (__bf16*)wsalloc((size_t)256 * 256 * 2);
  __bf16* CT3 = (__bf16*)wsalloc((size_t)256 * 256 * 2);
  __bf16* LT1 = (__bf16*)wsalloc((size_t)512 * 256 * 2);
  __bf16* LT2 = (__bf16*)wsalloc((size_t)256 * 256 * 2);
  __bf16* LT3 = (__bf16*)wsalloc((size_t)256 * 256 * 2);
  __bf16* VT1 = (__bf16*)wsalloc((size_t)512 * 256 * 2);
  __bf16* VT2 = (__bf16*)wsalloc((size_t)256 * 256 * 2);
  if (off > ws_size) return;  // workspace too small: bail deterministically

  // --- weight conversion (transpose to [N][K] bf16) ---
  auto conv = [&](int idx, __bf16* dst, int K, int N) {
    long n = (long)K * N;
    convert_wt_kernel<<<(unsigned)((n + 255) / 256), 256, 0, stream>>>(
        (const float*)d_in[idx], dst, K, N);
  };
  conv(iCW[0], CT1, 512, 256); conv(iCW[1], CT2, 256, 256); conv(iCW[2], CT3, 256, 256);
  conv(iLW[0], LT1, 512, 256); conv(iLW[1], LT2, 256, 256); conv(iLW[2], LT3, 256, 256);
  conv(iVW[0], VT1, 512, 256); conv(iVW[1], VT2, 256, 256);

  // --- state init ---
  {
    long nl = (long)N_LITS * FDIM, nc = (long)N_CLAUSES * FDIM;
    init_state_kernel<<<(unsigned)((nl + 255) / 256), 256, 0, stream>>>(
        lits, (const float*)d_in[iLinit], N_LITS);
    init_state_kernel<<<(unsigned)((nc + 255) / 256), 256, 0, stream>>>(
        cls, (const float*)d_in[iCinit], N_CLAUSES);
  }

  const long nClauseF4 = (long)N_CLAUSES * FDIM / 4;
  const long nLitF4 = (long)N_LITS * FDIM / 4;
  const unsigned scatterBlocks = (unsigned)(((long)nnz * 64 + 255) / 256);
  const unsigned clauseBlocks = (unsigned)(((N_CLAUSES + 15) / 16 + 3) / 4);
  const unsigned litBlocks = (unsigned)(((N_LITS + 15) / 16 + 3) / 4);
  const unsigned voteBlocks = (unsigned)(((N_VARS + 15) / 16 + 3) / 4);

  for (int r = 0; r < ROUNDS; ++r) {
    // clauses_new = segment_sum(lits[lit_idx] -> clause_idx); cls = LN(MLP([new|cls]))
    zero_f32_kernel<<<(unsigned)((nClauseF4 + 255) / 256), 256, 0, stream>>>(
        (float4*)msg, nClauseF4);
    scatter_add_kernel<<<scatterBlocks, 256, 0, stream>>>(lits, lit_idx, msg, clause_idx, nnz);
    mlp_ln_kernel<<<clauseBlocks, 128, 0, stream>>>(
        msg, cls, CT1, (const float*)d_in[iCb[0]], CT2, (const float*)d_in[iCb[1]],
        CT3, (const float*)d_in[iCb[2]], (const float*)d_in[iCg], (const float*)d_in[iCbe],
        cls, N_CLAUSES);
    // lits_new = segment_sum(cls[clause_idx] -> lit_idx); lits = LN(MLP([new|lits]))
    zero_f32_kernel<<<(unsigned)((nLitF4 + 255) / 256), 256, 0, stream>>>(
        (float4*)msg, nLitF4);
    scatter_add_kernel<<<scatterBlocks, 256, 0, stream>>>(cls, clause_idx, msg, lit_idx, nnz);
    mlp_ln_kernel<<<litBlocks, 128, 0, stream>>>(
        msg, lits, LT1, (const float*)d_in[iLb[0]], LT2, (const float*)d_in[iLb[1]],
        LT3, (const float*)d_in[iLb[2]], (const float*)d_in[iLg], (const float*)d_in[iLbe],
        lits, N_LITS);
  }

  vote_kernel<<<voteBlocks, 128, 0, stream>>>(
      lits, VT1, (const float*)d_in[iVb[0]], VT2, (const float*)d_in[iVb[1]],
      (const float*)d_in[iVW[2]], (const float*)d_in[iVb[2]], (float*)d_out, N_VARS);
}
